// LocalPoolPointnet_31739808317547
// MI455X (gfx1250) — matrix operand order repaired and checked by reading, not compiled, not run on hardware
//
#include <hip/hip_runtime.h>

// ---------------- problem constants ----------------
#define H    32        // hidden dim
#define H2   64        // 2*H
#define CDIM 32        // c_dim
#define MVOX 131072    // num voxels (reference constant)
#define NBLK 5
#define ETILE 128      // edges per workgroup (8 waves x 16 rows)
#define NFRAG 52       // 5 blocks * 10 fragments + 2 fc_c fragments

typedef __attribute__((ext_vector_type(16))) _Float16 v16h;
typedef __attribute__((ext_vector_type(8)))  float    v8f;

// ---------------- async global->LDS copy (CDNA5 path) ----------------
// GLOBAL_LOAD_ASYNC_TO_LDS_B128 (VGLOBAL op 98): VDST = LDS byte address,
// VADDR = 64-bit global address, tracked by ASYNCcnt.  Issued via inline asm
// (clang builtin signature differs across toolchains).
#define HAVE_ASYNC_LDS 1

// copy 16 bytes global -> LDS without VGPR staging
static __device__ __forceinline__ void copy16_g2l(const void* g, void* l) {
#if HAVE_ASYNC_LDS
    // low 32 bits of a generic LDS pointer are the LDS byte address
    unsigned lds_addr = (unsigned)(unsigned long long)l;
    unsigned long long gaddr = (unsigned long long)g;
    asm volatile("global_load_async_to_lds_b128 %0, %1, off"
                 :
                 : "v"(lds_addr), "v"(gaddr)
                 : "memory");
#else
    *(uint4*)l = *(const uint4*)g;
#endif
}

static __device__ __forceinline__ void async_wait() {
#if HAVE_ASYNC_LDS
    asm volatile("s_wait_asynccnt 0" ::: "memory");
#endif
}

static __device__ __forceinline__ v8f wmma16(v16h a, v16h b, v8f c) {
    // D = A(16x32 f16) * B(32x16 f16) + C(16x16 f32)
    return __builtin_amdgcn_wmma_f32_16x16x32_f16(
        false, a, false, b, (short)0, c, false, false);
}

// A fragment: 16x32 slice (rows m=0..15, cols k0..k0+31) of an LDS f16 tile
// [16][ld].  ISA 16-bit A layout: lanes 0-15 hold K {0..7,16..23}, lanes
// 16-31 hold K {8..15,24..31} of the 32-wide slice.
static __device__ __forceinline__ v16h load_a(const _Float16* t, int ld, int k0,
                                              int lane, bool relu) {
    v16h a;
    const int m  = lane & 15;
    const int kb = k0 + ((lane & 16) ? 8 : 0);
#pragma unroll
    for (int e = 0; e < 16; ++e) {
        int k = kb + e + ((e < 8) ? 0 : 8);
        _Float16 v = t[m * ld + k];
        if (relu) {
            float fv = (float)v;
            v = (fv > 0.f) ? v : (_Float16)0.f;
        }
        a[e] = v;
    }
    return a;
}

// Pre-packed B fragment: one coalesced 32B load per lane.
static __device__ __forceinline__ v16h load_bp(const _Float16* __restrict__ pack,
                                               int frag, int lane) {
    return *(const v16h*)(pack + ((size_t)frag << 9) + (lane << 4));
}

// One ConvONet ResnetBlockFC (2H -> H) for this wave's 16 edge rows.
//   net  = relu(x) @ W0 + b0
//   net2 = relu(net) @ W1 + b1
//   out  = x @ Ws + net2
// xt: wave's 16x64 f16 input tile in LDS; nt: wave's 16x32 f16 scratch.
// pk:  this block's 10 pre-packed B fragments:
//      [0..3]=W0(kb,nb) [4..5]=W1(nb) [6..9]=Ws(kb,nb)
static __device__ __forceinline__ void resnet_wave(
    const _Float16* xt, _Float16* nt, const _Float16* __restrict__ pk,
    const float* __restrict__ b0, const float* __restrict__ b1,
    int lane, v8f out[2]) {
    v8f acc0, acc1;
    {
        float c0 = b0[lane & 15];
        float c1 = b0[16 + (lane & 15)];
#pragma unroll
        for (int r = 0; r < 8; ++r) { acc0[r] = c0; acc1[r] = c1; }
    }
#pragma unroll
    for (int kbk = 0; kbk < 2; ++kbk) {
        v16h a = load_a(xt, H2, kbk * 32, lane, true);          // relu(x)
        acc0 = wmma16(a, load_bp(pk, kbk * 2 + 0, lane), acc0);
        acc1 = wmma16(a, load_bp(pk, kbk * 2 + 1, lane), acc1);
    }
    // stash relu(net) in LDS so it can be reloaded in A-fragment layout
#pragma unroll
    for (int r = 0; r < 8; ++r) {
        int m = r + ((lane & 16) ? 8 : 0);
        float v0 = acc0[r] > 0.f ? acc0[r] : 0.f;
        float v1 = acc1[r] > 0.f ? acc1[r] : 0.f;
        nt[m * H + (lane & 15)]      = (_Float16)v0;
        nt[m * H + 16 + (lane & 15)] = (_Float16)v1;
    }
    __syncthreads();
    v8f o0, o1;
    {
        float c0 = b1[lane & 15];
        float c1 = b1[16 + (lane & 15)];
#pragma unroll
        for (int r = 0; r < 8; ++r) { o0[r] = c0; o1[r] = c1; }
    }
    v16h an = load_a(nt, H, 0, lane, false);                    // relu(net)
    o0 = wmma16(an, load_bp(pk, 4, lane), o0);
    o1 = wmma16(an, load_bp(pk, 5, lane), o1);
#pragma unroll
    for (int kbk = 0; kbk < 2; ++kbk) {
        v16h ax = load_a(xt, H2, kbk * 32, lane, false);        // raw x shortcut
        o0 = wmma16(ax, load_bp(pk, 6 + kbk * 2 + 0, lane), o0);
        o1 = wmma16(ax, load_bp(pk, 6 + kbk * 2 + 1, lane), o1);
    }
    out[0] = o0; out[1] = o1;
}

// store this wave's 16x32 f32 result to the global f16 feat buffer
static __device__ __forceinline__ void store_feat(const v8f out[2],
                                                  _Float16* __restrict__ feat,
                                                  int base, int wave, int lane,
                                                  int E) {
#pragma unroll
    for (int nb = 0; nb < 2; ++nb)
#pragma unroll
        for (int r = 0; r < 8; ++r) {
            int m = r + ((lane & 16) ? 8 : 0);
            int e = base + wave * 16 + m;
            if (e < E)
                feat[(size_t)e * H + nb * 16 + (lane & 15)] = (_Float16)out[nb][r];
        }
}

// ---------------- kernels ----------------

// Pre-pack all WMMA B fragments (f32 weights -> per-lane f16 fragment layout).
// Fragment g, lane l, element e lives at pack[g*512 + l*16 + e]; B layout:
// lanes 0-15 hold K k0+{0..7,16..23}, lanes 16-31 hold K k0+{8..15,24..31}; N = n0+(l&15).
__global__ void k_pack_w(const float* __restrict__ W0, const float* __restrict__ W1,
                         const float* __restrict__ Ws, const float* __restrict__ Wc,
                         _Float16* __restrict__ pack) {
    int idx = blockIdx.x * blockDim.x + threadIdx.x;
    if (idx >= NFRAG * 512) return;
    int g    = idx >> 9;
    int lane = (idx >> 4) & 31;
    int e    = idx & 15;
    const float* W; int k0, n0;
    if (g < 50) {
        int i = g / 10, j = g % 10;
        if (j < 4)      { W = W0 + (size_t)i * H2 * H; k0 = (j >> 1) * 32; n0 = (j & 1) * 16; }
        else if (j < 6) { W = W1 + (size_t)i * H  * H; k0 = 0;             n0 = (j & 1) * 16; }
        else { j -= 6;    W = Ws + (size_t)i * H2 * H; k0 = (j >> 1) * 32; n0 = (j & 1) * 16; }
    } else {
        W = Wc; k0 = 0; n0 = (g & 1) * 16;
    }
    int K = k0 + ((lane & 16) ? 8 : 0) + e + ((e < 8) ? 0 : 8);
    int N = n0 + (lane & 15);
    pack[idx] = (_Float16)W[K * H + N];
}

// gather + fc_pos + resnet block 0
__global__ __launch_bounds__(256) void k_block0(
    const float* __restrict__ xyz, const int* __restrict__ pids,
    const float* __restrict__ Wp, const float* __restrict__ bp,
    const _Float16* __restrict__ pk,
    const float* __restrict__ b0, const float* __restrict__ b1,
    _Float16* __restrict__ feat, int E) {
    __shared__ _Float16 xt[ETILE * H2];
    __shared__ _Float16 nt[ETILE * H];
    const int tid  = threadIdx.x;
    const int base = blockIdx.x * ETILE;
    {   // fc_pos: each thread fills 32 columns of one row
        int r = tid >> 1, h = tid & 1;
        int e = base + r; if (e >= E) e = E - 1;
        int p = pids[e];
        float x0 = xyz[p * 3 + 0], x1 = xyz[p * 3 + 1], x2 = xyz[p * 3 + 2];
#pragma unroll
        for (int c = 0; c < 32; ++c) {
            int col = h * 32 + c;
            float f = bp[col] + x0 * Wp[col] + x1 * Wp[H2 + col] + x2 * Wp[2 * H2 + col];
            xt[r * H2 + col] = (_Float16)f;
        }
    }
    __syncthreads();
    const int wave = tid >> 5, lane = tid & 31;
    v8f out[2];
    resnet_wave(xt + wave * 16 * H2, nt + wave * 16 * H, pk, b0, b1, lane, out);
    store_feat(out, feat, base, wave, lane, E);
}

// seed pooled-max buffer with encode(-inf)
__global__ void k_pool_init(unsigned* __restrict__ pool, int n) {
    int i = blockIdx.x * blockDim.x + threadIdx.x;
    if (i < n) pool[i] = 0x007FFFFFu;   // monotone encoding of -inf
}

// segment max via order-preserving uint encoding + atomicMax
__global__ void k_segmax(const _Float16* __restrict__ feat,
                         const int* __restrict__ vids,
                         unsigned* __restrict__ pool, int E) {
    int i = blockIdx.x * blockDim.x + threadIdx.x;
    if (i >= E * H) return;
    int e = i >> 5, j = i & 31;
    float v = (float)feat[i];
    unsigned b = __float_as_uint(v);
    unsigned enc = (b & 0x80000000u) ? ~b : (b | 0x80000000u);
    atomicMax(&pool[(size_t)vids[e] * H + j], enc);
}

// decode encoded max, map empty segments (-inf) to 0, store f16
__global__ void k_decode(const unsigned* __restrict__ pool,
                         _Float16* __restrict__ ph, int n) {
    int i = blockIdx.x * blockDim.x + threadIdx.x;
    if (i >= n) return;
    unsigned u = pool[i];
    float v = 0.f;
    if (u != 0x007FFFFFu) {
        unsigned b = (u & 0x80000000u) ? (u & 0x7FFFFFFFu) : ~u;
        v = __uint_as_float(b);
    }
    ph[i] = (_Float16)v;
}

// concat(feat, pooled[vids]) + resnet block i (in-place feat update)
__global__ __launch_bounds__(256) void k_block_mid(
    const int* __restrict__ vids, const _Float16* __restrict__ pooled,
    const _Float16* __restrict__ pk,
    const float* __restrict__ b0, const float* __restrict__ b1,
    _Float16* __restrict__ feat, int E) {
    __shared__ _Float16 xt[ETILE * H2];
    __shared__ _Float16 nt[ETILE * H];
    const int tid  = threadIdx.x;
    const int base = blockIdx.x * ETILE;
    {   // each thread stages one 32-half half-row (64B) into the concat tile
        int r = tid >> 1, h = tid & 1;
        int e = base + r; int ec = (e < E) ? e : (E - 1);
        const _Float16* srcp = (h == 0)
            ? feat   + (size_t)ec * H
            : pooled + (size_t)vids[ec] * H;
        _Float16* dstp = xt + r * H2 + h * 32;
#pragma unroll
        for (int q = 0; q < 4; ++q)
            copy16_g2l(srcp + q * 8, dstp + q * 8);   // async global->LDS
    }
    async_wait();
    __syncthreads();
    const int wave = tid >> 5, lane = tid & 31;
    v8f out[2];
    resnet_wave(xt + wave * 16 * H2, nt + wave * 16 * H, pk, b0, b1, lane, out);
    store_feat(out, feat, base, wave, lane, E);
}

__global__ void k_zero(float* __restrict__ csum, int ncs,
                       float* __restrict__ cnt, int nc) {
    int i = blockIdx.x * blockDim.x + threadIdx.x;
    if (i < ncs) csum[i] = 0.f;
    if (i < nc)  cnt[i]  = 0.f;
}

// fc_c (WMMA 32x32) + segment-sum + count
__global__ __launch_bounds__(256) void k_fcc(
    const _Float16* __restrict__ feat, const int* __restrict__ vids,
    const _Float16* __restrict__ pk,   // 2 fragments of fc_c_w
    const float* __restrict__ bc,
    float* __restrict__ csum, float* __restrict__ cnt, int E) {
    __shared__ _Float16 xt[ETILE * H];
    const int tid  = threadIdx.x;
    const int base = blockIdx.x * ETILE;
    if (tid < ETILE) {
        int e = base + tid; int ec = (e < E) ? e : (E - 1);
        const _Float16* srcp = feat + (size_t)ec * H;
        _Float16* dstp = xt + tid * H;
#pragma unroll
        for (int q = 0; q < 4; ++q)
            copy16_g2l(srcp + q * 8, dstp + q * 8);   // async global->LDS
    }
    async_wait();
    __syncthreads();
    const int wave = tid >> 5, lane = tid & 31;
    const _Float16* xw = xt + wave * 16 * H;
    v8f a0, a1;
    {
        float c0 = bc[lane & 15];
        float c1 = bc[16 + (lane & 15)];
#pragma unroll
        for (int r = 0; r < 8; ++r) { a0[r] = c0; a1[r] = c1; }
    }
    v16h af = load_a(xw, H, 0, lane, false);
    a0 = wmma16(af, load_bp(pk, 0, lane), a0);
    a1 = wmma16(af, load_bp(pk, 1, lane), a1);
#pragma unroll
    for (int r = 0; r < 8; ++r) {
        int m = r + ((lane & 16) ? 8 : 0);
        int e = base + wave * 16 + m;
        if (e < E) {
            int v = vids[e];
            atomicAdd(&csum[(size_t)v * CDIM + (lane & 15)],      a0[r]);
            atomicAdd(&csum[(size_t)v * CDIM + 16 + (lane & 15)], a1[r]);
            if ((lane & 15) == 0) atomicAdd(&cnt[v], 1.0f);  // one per row
        }
    }
}

__global__ void k_final(const float* __restrict__ csum,
                        const float* __restrict__ cnt,
                        float* __restrict__ out, int n) {
    int i = blockIdx.x * blockDim.x + threadIdx.x;
    if (i >= n) return;
    float c = cnt[i >> 5];
    out[i] = csum[i] / (c > 1.f ? c : 1.f);
}

// ---------------- launcher ----------------
extern "C" void kernel_launch(void* const* d_in, const int* in_sizes, int n_in,
                              void* d_out, int out_size, void* d_ws, size_t ws_size,
                              hipStream_t stream) {
    (void)n_in; (void)out_size; (void)ws_size;
    const float* xyz  = (const float*)d_in[0];
    const int*   pids = (const int*)d_in[1];
    const int*   vids = (const int*)d_in[2];
    // d_in[3] = num_voxels (device scalar) -- fixed at MVOX per the reference
    const float* Wp = (const float*)d_in[4];
    const float* bp = (const float*)d_in[5];
    const float* W0 = (const float*)d_in[6];
    const float* b0 = (const float*)d_in[7];
    const float* W1 = (const float*)d_in[8];
    const float* b1 = (const float*)d_in[9];
    const float* Ws = (const float*)d_in[10];
    const float* Wc = (const float*)d_in[11];
    const float* bc = (const float*)d_in[12];
    const int E = in_sizes[1];
    const int M = MVOX;

    // workspace: feat f16 | pool u32 | pooled f16 | csum f32 | cnt f32 | pack f16
    char* ws = (char*)d_ws;
    size_t o = 0;
    _Float16* feat = (_Float16*)(ws + o); o += (size_t)E * H * 2;    o = (o + 255) & ~(size_t)255;
    unsigned* pool = (unsigned*)(ws + o); o += (size_t)M * H * 4;    o = (o + 255) & ~(size_t)255;
    _Float16* ph   = (_Float16*)(ws + o); o += (size_t)M * H * 2;    o = (o + 255) & ~(size_t)255;
    float*    csum = (float*)(ws + o);    o += (size_t)M * CDIM * 4; o = (o + 255) & ~(size_t)255;
    float*    cnt  = (float*)(ws + o);    o += (size_t)M * 4;        o = (o + 255) & ~(size_t)255;
    _Float16* pack = (_Float16*)(ws + o);

    const int npk = NFRAG * 512;
    k_pack_w<<<(npk + 255) / 256, 256, 0, stream>>>(W0, W1, Ws, Wc, pack);

    const int tiles = (E + ETILE - 1) / ETILE;
    k_block0<<<tiles, 256, 0, stream>>>(xyz, pids, Wp, bp, pack, b0, b1, feat, E);

    for (int i = 1; i < NBLK; ++i) {
        const int npool = M * H;
        k_pool_init<<<(npool + 255) / 256, 256, 0, stream>>>(pool, npool);
        const int nfeat = E * H;
        k_segmax<<<(nfeat + 255) / 256, 256, 0, stream>>>(feat, vids, pool, E);
        k_decode<<<(npool + 255) / 256, 256, 0, stream>>>(pool, ph, npool);
        k_block_mid<<<tiles, 256, 0, stream>>>(vids, ph,
                                               pack + (size_t)i * 10 * 512,
                                               b0 + (size_t)i * H, b1 + (size_t)i * H,
                                               feat, E);
    }

    const int ncs = M * CDIM;
    k_zero<<<(ncs + 255) / 256, 256, 0, stream>>>(csum, ncs, cnt, M);
    k_fcc<<<tiles, 256, 0, stream>>>(feat, vids, pack + (size_t)50 * 512, bc, csum, cnt, E);
    k_final<<<(ncs + 255) / 256, 256, 0, stream>>>(csum, cnt, (float*)d_out, ncs);
}
